// MHA_19138374271206
// MI455X (gfx1250) — compile-verified, matrix operand order checked by reading, not run on hardware
//
#include <hip/hip_runtime.h>

#define S_LEN 1024
#define B_SZ  4
#define D_DIM 1024
#define H_CNT 16
#define DHEAD 64
#define SB    (S_LEN * B_SZ)   // 4096 rows

typedef __attribute__((ext_vector_type(16))) __bf16   v16bf;
typedef __attribute__((ext_vector_type(8)))  __bf16   v8bf;
typedef __attribute__((ext_vector_type(8)))  float    v8f;
typedef __attribute__((ext_vector_type(4)))  unsigned v4u;
typedef __attribute__((ext_vector_type(8)))  int      v8i;
typedef __attribute__((ext_vector_type(4)))  int      v4i;

#if __has_builtin(__builtin_amdgcn_tensor_load_to_lds) && \
    __has_builtin(__builtin_amdgcn_s_wait_tensorcnt)
#define USE_TDM 1
#else
#define USE_TDM 0
#endif

// ---- fragment load: two contiguous 16B chunks -> 16x32 bf16 A/B fragment ----
// lanes 0-15 hold K 0..7 / 16..23, lanes 16-31 hold K 8..15 / 24..31 (ISA 7.12.2)
__device__ __forceinline__ v16bf frag_ld(const __bf16* rowp, int cbase, int hi) {
    v8bf c0 = *(const v8bf*)(rowp + cbase + hi * 8);
    v8bf c1 = *(const v8bf*)(rowp + cbase + 16 + hi * 8);
    return __builtin_shufflevector(c0, c1, 0, 1, 2, 3, 4, 5, 6, 7,
                                   8, 9, 10, 11, 12, 13, 14, 15);
}

__device__ __forceinline__ float rmax16(float x) {
#pragma unroll
    for (int o = 1; o < 16; o <<= 1) x = fmaxf(x, __shfl_xor(x, o, 16));
    return x;
}

__device__ __forceinline__ v8f wmma_bf16(v16bf a, v16bf b, v8f c) {
    return __builtin_amdgcn_wmma_f32_16x16x32_bf16(false, a, false, b,
                                                   (short)0, c, false, false);
}

// ---- TDM: 2D tile load, D# per ISA 8.3-8.5 (data_size=2B, type=2) ----------
__device__ __forceinline__ void tdm_load_2d(unsigned lds_off, const void* gaddr,
                                            unsigned tile_d0, unsigned tile_d1,
                                            unsigned tens_d0, unsigned tens_d1,
                                            unsigned stride0) {
#if USE_TDM
    unsigned long long ga = (unsigned long long)gaddr;
    v4u g0;
    g0[0] = 1u;                                  // count=1, user mode
    g0[1] = lds_off;                             // lds_addr (bytes)
    g0[2] = (unsigned)ga;                        // global_addr[31:0]
    g0[3] = ((unsigned)(ga >> 32) & 0x01FFFFFFu) | 0x80000000u; // addr[56:32]|type=2
    v8i g1;
    g1[0] = 0x10000;                             // wg_mask=0, data_size=1 (2B)
    g1[1] = (int)((tens_d0 & 0xFFFFu) << 16);    // tensor_dim0[15:0]
    g1[2] = (int)((tens_d0 >> 16) | ((tens_d1 & 0xFFFFu) << 16));
    g1[3] = (int)((tens_d1 >> 16) | ((tile_d0 & 0xFFFFu) << 16));
    g1[4] = (int)(tile_d1 & 0xFFFFu);            // tile_dim1, tile_dim2=0
    g1[5] = (int)stride0;                        // tensor_dim0_stride[31:0]
    g1[6] = 0;
    g1[7] = 0;
    v4i z4 = {0, 0, 0, 0};
#if __clang_major__ >= 23
    v8i z8 = {0, 0, 0, 0, 0, 0, 0, 0};
    __builtin_amdgcn_tensor_load_to_lds(g0, g1, z4, z4, z8, 0);
#else
    __builtin_amdgcn_tensor_load_to_lds(g0, g1, z4, z4, 0);
#endif
#endif
}

// ------------------------------------------------------------ f32 -> bf16 cvt
__global__ void cvt_f32_bf16(const float* __restrict__ src, __bf16* __restrict__ dst,
                             unsigned n) {
    unsigned i = (blockIdx.x * blockDim.x + threadIdx.x) * 8u;
    if (i >= n) return;
    v8f f = *(const v8f*)(src + i);
    v8bf o;
#pragma unroll
    for (int t = 0; t < 8; ++t) o[t] = (__bf16)f[t];
    *(v8bf*)(dst + i) = o;
}

__global__ void cvt_weights(const float* __restrict__ w0, const float* __restrict__ w1,
                            const float* __restrict__ w2, const float* __restrict__ w3,
                            __bf16* __restrict__ dst) {
    unsigned i = (blockIdx.x * blockDim.x + threadIdx.x) * 8u;
    const unsigned n = D_DIM * D_DIM;
    if (i >= 4u * n) return;
    const float* src = (i < n) ? w0 : (i < 2 * n) ? w1 : (i < 3 * n) ? w2 : w3;
    v8f f = *(const v8f*)(src + (i & (n - 1)));
    v8bf o;
#pragma unroll
    for (int t = 0; t < 8; ++t) o[t] = (__bf16)f[t];
    *(v8bf*)(dst + i) = o;
}

// --------------------------------------------------------- unified bf16 GEMM
// out[r,n] = sum_d X[r,d]*W[n,d] + bias[n]; 2 M-tiles x 4 N-tiles per wave.
// OMODE 0: bf16 (H,B,S,DH) | OMODE 1: bf16 (H,B,DH,S) | OMODE 2: f32 (S,B,D)
template <int OMODE>
__global__ __launch_bounds__(128) void gemm_bf16(
    const __bf16* __restrict__ X, const __bf16* __restrict__ W,
    const float* __restrict__ bias, void* __restrict__ outp) {
    int wave = threadIdx.x >> 5, lane = threadIdx.x & 31;
    int tile = blockIdx.x * 4 + wave;           // 2048 waves
    int tmg = tile >> 4;                        // 0..127 (32-row groups)
    int tng = tile & 15;                        // 0..15  (64-col groups)
    int m = lane & 15, hi = lane >> 4;
    int rm = tmg * 32, cn = tng * 64;

    const __bf16* Xr0 = X + (size_t)(rm + m) * D_DIM;
    const __bf16* Xr1 = X + (size_t)(rm + 16 + m) * D_DIM;
    const __bf16* Wr0 = W + (size_t)(cn + m) * D_DIM;
    const __bf16* Wr1 = W + (size_t)(cn + 16 + m) * D_DIM;
    const __bf16* Wr2 = W + (size_t)(cn + 32 + m) * D_DIM;
    const __bf16* Wr3 = W + (size_t)(cn + 48 + m) * D_DIM;

    v8f acc[2][4] = {};
    for (int kk = 0; kk < D_DIM; kk += 32) {
        v16bf a0 = frag_ld(Xr0, kk, hi);
        v16bf a1 = frag_ld(Xr1, kk, hi);
        v16bf b0 = frag_ld(Wr0, kk, hi);
        v16bf b1 = frag_ld(Wr1, kk, hi);
        v16bf b2 = frag_ld(Wr2, kk, hi);
        v16bf b3 = frag_ld(Wr3, kk, hi);
        acc[0][0] = wmma_bf16(a0, b0, acc[0][0]);
        acc[0][1] = wmma_bf16(a0, b1, acc[0][1]);
        acc[0][2] = wmma_bf16(a0, b2, acc[0][2]);
        acc[0][3] = wmma_bf16(a0, b3, acc[0][3]);
        acc[1][0] = wmma_bf16(a1, b0, acc[1][0]);
        acc[1][1] = wmma_bf16(a1, b1, acc[1][1]);
        acc[1][2] = wmma_bf16(a1, b2, acc[1][2]);
        acc[1][3] = wmma_bf16(a1, b3, acc[1][3]);
    }

    if constexpr (OMODE == 1) {
        // Transposed-V epilogue: wave tile = one head, rows = 8 s x 4 b.
        // Stage [dh][b][s'] in LDS, then 16B contiguous stores along s.
        __shared__ __attribute__((aligned(16))) __bf16 tb[4][2048];  // 4KB/wave
        __bf16* tw = tb[wave];
        int hh = cn >> 6;                    // head index (cn multiple of 64)
        int s0 = rm >> 2;                    // first s of this tile
#pragma unroll
        for (int j = 0; j < 4; ++j) {
            int dh = j * 16 + m;
            float bv = bias[cn + j * 16 + m];
#pragma unroll
            for (int mi = 0; mi < 2; ++mi)
#pragma unroll
                for (int i = 0; i < 8; ++i) {
                    int r = mi * 16 + i + hi * 8;          // 0..31
                    int sp = r >> 2, bb = r & 3;
                    tw[dh * 32 + bb * 8 + sp] = (__bf16)(acc[mi][j][i] + bv);
                }
        }
        asm volatile("s_wait_dscnt 0" ::: "memory");       // wave-local RAW
        __bf16* vout = (__bf16*)outp;
#pragma unroll
        for (int c = 0; c < 8; ++c) {
            int chunk = lane * 8 + c;                      // 256 x 16B chunks
            int dh = chunk >> 2, bb = chunk & 3;
            *(v8bf*)(vout + ((size_t)(hh * B_SZ + bb) * DHEAD + dh) * S_LEN + s0) =
                *(const v8bf*)(tw + dh * 32 + bb * 8);
        }
    } else {
#pragma unroll
        for (int j = 0; j < 4; ++j) {
            int col = cn + j * 16 + m;
            float bv = bias[col];
            int hh = col >> 6, dh = col & 63;
#pragma unroll
            for (int mi = 0; mi < 2; ++mi)
#pragma unroll
                for (int i = 0; i < 8; ++i) {
                    int row = rm + mi * 16 + i + hi * 8;   // row = s*B + b
                    int s = row >> 2, bb = row & 3;
                    float val = acc[mi][j][i] + bv;
                    if (OMODE == 0)
                        ((__bf16*)outp)[((size_t)(hh * B_SZ + bb) * S_LEN + s) * DHEAD + dh] =
                            (__bf16)val;
                    else
                        ((float*)outp)[(size_t)row * D_DIM + col] = val;
                }
        }
    }
}

// ------------------------------------------------------------- flash attention
// block = 4 waves sharing one (h,b); each wave owns a 16-row query block.
// K tile 32x64 and V(transposed) tile 64x32 staged in LDS (TDM, double-buffered)
#define KT_OFF(buf) ((buf) * 4096u)
#define VT_OFF(buf) (8192u + (buf) * 4096u)
#define P_OFF       16384u
#define SMEM_BYTES  20480u

__global__ __launch_bounds__(128) void flash_attn(
    const __bf16* __restrict__ Qh, const __bf16* __restrict__ Kh,
    const __bf16* __restrict__ Vt, const float* __restrict__ mask,
    __bf16* __restrict__ Ctx) {
    __shared__ __attribute__((aligned(16))) char smem[SMEM_BYTES];

    int wave = threadIdx.x >> 5, lane = threadIdx.x & 31;
    int hb = blockIdx.x >> 4;                // 0..63 = h*B + b
    int qg = blockIdx.x & 15;
    int b = hb & 3, h = hb >> 2;
    size_t base = (size_t)hb * S_LEN * DHEAD;
    int m = lane & 15, hi = lane >> 4;
    int q0 = (qg * 4 + wave) * 16;
    const float scale = 0.125f;              // 1/sqrt(64)

    const __bf16* Kg = Kh + base;
    const __bf16* Vg = Vt + base;
    __bf16* Pw = (__bf16*)(smem + P_OFF) + wave * 512;   // private 16x32 tile

    auto stage = [&](int buf, int kg) {
#if USE_TDM
        if (threadIdx.x < 32) {              // one wave issues both DMAs
            tdm_load_2d(KT_OFF(buf), Kg + (size_t)kg * DHEAD,
                        64, 32, DHEAD, S_LEN, DHEAD);
            tdm_load_2d(VT_OFF(buf), Vg + kg,
                        32, 64, S_LEN, DHEAD, S_LEN);
        }
#else
        {   // cooperative copy fallback
            __bf16* kd = (__bf16*)(smem + KT_OFF(buf));
            __bf16* vd = (__bf16*)(smem + VT_OFF(buf));
            const __bf16* ks = Kg + (size_t)kg * DHEAD;  // contiguous 2048 elems
#pragma unroll
            for (int c = 0; c < 2; ++c) {
                int idx = (threadIdx.x * 2 + c) * 8;     // 256 x v8bf
                *(v8bf*)(kd + idx) = *(const v8bf*)(ks + idx);
                int r = (threadIdx.x * 2 + c) >> 2;      // 64 rows x 4 chunks
                int p = ((threadIdx.x * 2 + c) & 3) * 8;
                *(v8bf*)(vd + r * 32 + p) =
                    *(const v8bf*)(Vg + (size_t)r * S_LEN + kg + p);
            }
        }
#endif
    };
    auto settle = [&]() {
        asm volatile("s_wait_dscnt 0" ::: "memory");
#if USE_TDM
        if (threadIdx.x < 32) __builtin_amdgcn_s_wait_tensorcnt(0);
#endif
        __syncthreads();
    };

    // Q fragments: 16x64 -> two 16x32 A tiles (b128 global loads)
    const __bf16* Qr = Qh + base + (size_t)(q0 + m) * DHEAD;
    v16bf qf0 = frag_ld(Qr, 0, hi);
    v16bf qf1 = frag_ld(Qr, 32, hi);
    const float* mrow = mask + (size_t)(q0 + hi * 8) * S_LEN;

    v16bf onesB;                             // all-ones B: row-sum via WMMA
#pragma unroll
    for (int t = 0; t < 16; ++t) onesB[t] = (__bf16)1.0f;

    v8f acc[4] = {};
    float mI[8], lI[8];
#pragma unroll
    for (int i = 0; i < 8; ++i) { mI[i] = -3.0e38f; lI[i] = 0.0f; }

    stage(0, 0);
    settle();

    for (int kt = 0; kt < S_LEN / 32; ++kt) {
        int cur = kt & 1;
        int kg = kt * 32;
        if (kt + 1 < S_LEN / 32) stage(cur ^ 1, kg + 32);

        const __bf16* Kt = (const __bf16*)(smem + KT_OFF(cur));
        const __bf16* Vl = (const __bf16*)(smem + VT_OFF(cur));

        // scores: two 16x16 C tiles via 4 WMMAs (ds_load_b128 fragments)
        v8f s0 = {}, s1 = {};
#pragma unroll
        for (int c = 0; c < 2; ++c) {
            v16bf kb0 = frag_ld(Kt + m * DHEAD, c * 32, hi);
            v16bf kb1 = frag_ld(Kt + (16 + m) * DHEAD, c * 32, hi);
            s0 = wmma_bf16(c ? qf1 : qf0, kb0, s0);
            s1 = wmma_bf16(c ? qf1 : qf0, kb1, s1);
        }

        // online softmax: row max via lane shuffles, row sum via ones-WMMA
        float alphaV[8];
#pragma unroll
        for (int i = 0; i < 8; ++i) {
            float e0 = s0[i] * scale + mrow[(size_t)i * S_LEN + kg + m];
            float e1 = s1[i] * scale + mrow[(size_t)i * S_LEN + kg + 16 + m];
            float tmax = rmax16(fmaxf(e0, e1));
            float mnew = fmaxf(mI[i], tmax);
            alphaV[i] = __expf(mI[i] - mnew);
            mI[i] = mnew;
            Pw[(i + hi * 8) * 32 + m]      = (__bf16)__expf(e0 - mnew);
            Pw[(i + hi * 8) * 32 + 16 + m] = (__bf16)__expf(e1 - mnew);
        }

        // P back as A-fragment: plain row-major b128 LDS reads (no transpose)
        v16bf pf = frag_ld(Pw + m * 32, 0, hi);

        // row sums of P, broadcast in C-layout (replaces 32 ds_bpermute)
        v8f ls = wmma_bf16(pf, onesB, v8f{});
#pragma unroll
        for (int i = 0; i < 8; ++i) {
            lI[i] = lI[i] * alphaV[i] + ls[i];
            acc[0][i] *= alphaV[i]; acc[1][i] *= alphaV[i];
            acc[2][i] *= alphaV[i]; acc[3][i] *= alphaV[i];
        }

        // ctx += P(16x32) x V(32x64); V fragments contiguous from Vt tile
#pragma unroll
        for (int dt = 0; dt < 4; ++dt) {
            v16bf vb = frag_ld(Vl + (dt * 16 + m) * 32, 0, hi);
            acc[dt] = wmma_bf16(pf, vb, acc[dt]);
        }

        settle();   // retire ds reads + next-tile DMA before buffer swap
    }

    // normalize, write ctx in (S,B,D) bf16
#pragma unroll
    for (int i = 0; i < 8; ++i) {
        float inv = 1.0f / lI[i];
        int row = q0 + i + hi * 8;
        size_t ob = ((size_t)row * B_SZ + b) * D_DIM + h * DHEAD;
#pragma unroll
        for (int dt = 0; dt < 4; ++dt)
            Ctx[ob + dt * 16 + m] = (__bf16)(acc[dt][i] * inv);
    }
}

// ----------------------------------------------------------------- launcher
extern "C" void kernel_launch(void* const* d_in, const int* in_sizes, int n_in,
                              void* d_out, int out_size, void* d_ws, size_t ws_size,
                              hipStream_t stream) {
    const float* q    = (const float*)d_in[0];
    const float* k    = (const float*)d_in[1];
    const float* v    = (const float*)d_in[2];
    const float* mask = (const float*)d_in[3];
    const float* Wq   = (const float*)d_in[4];
    const float* bq   = (const float*)d_in[5];
    const float* Wk   = (const float*)d_in[6];
    const float* bk   = (const float*)d_in[7];
    const float* Wv   = (const float*)d_in[8];
    const float* bv   = (const float*)d_in[9];
    const float* Wo   = (const float*)d_in[10];
    const float* bo   = (const float*)d_in[11];
    float* out = (float*)d_out;

    char* ws = (char*)d_ws;
    const size_t nDD  = (size_t)D_DIM * D_DIM;
    const size_t nSBD = (size_t)SB * D_DIM;
    __bf16* wq_bf = (__bf16*)ws;
    __bf16* wk_bf = wq_bf + nDD;
    __bf16* wv_bf = wq_bf + 2 * nDD;
    __bf16* wo_bf = wq_bf + 3 * nDD;
    __bf16* qh  = (__bf16*)(ws + 4 * nDD * sizeof(__bf16));
    __bf16* kh  = qh + nSBD;
    __bf16* vt  = qh + 2 * nSBD;      // transposed (H,B,DH,S)
    __bf16* ctx = qh + 3 * nSBD;
    __bf16* xq  = qh + 4 * nSBD;      // bf16 copies of q,k,v
    __bf16* xk  = qh + 5 * nSBD;
    __bf16* xv  = qh + 6 * nSBD;

    const unsigned cvtGrid = (unsigned)(nSBD / 8 / 256);
    cvt_f32_bf16<<<cvtGrid, 256, 0, stream>>>(q, xq, (unsigned)nSBD);
    cvt_f32_bf16<<<cvtGrid, 256, 0, stream>>>(k, xk, (unsigned)nSBD);
    cvt_f32_bf16<<<cvtGrid, 256, 0, stream>>>(v, xv, (unsigned)nSBD);
    cvt_weights<<<dim3((unsigned)(4 * nDD / 8 / 256)), 256, 0, stream>>>(
        Wq, Wk, Wv, Wo, wq_bf);

    dim3 gemmGrid((SB / 32) * (D_DIM / 64) / 4);      // 512 blocks x 4 waves
    gemm_bf16<0><<<gemmGrid, 128, 0, stream>>>(xq, wq_bf, bq, qh);
    gemm_bf16<0><<<gemmGrid, 128, 0, stream>>>(xk, wk_bf, bk, kh);
    gemm_bf16<1><<<gemmGrid, 128, 0, stream>>>(xv, wv_bf, bv, vt);

    dim3 attnGrid(64 * 16);           // (h,b) x 16 query groups, 4 waves each
    flash_attn<<<attnGrid, 128, 0, stream>>>(qh, kh, vt, mask, ctx);

    gemm_bf16<2><<<gemmGrid, 128, 0, stream>>>(ctx, wo_bf, bo, out);
}
